// Attention_65644280152150
// MI455X (gfx1250) — matrix operand order repaired and checked by reading, not compiled, not run on hardware
//
#include <hip/hip_runtime.h>

// ---------- types ----------
typedef __attribute__((ext_vector_type(16))) __bf16       v16bf;
typedef __attribute__((ext_vector_type(8)))  float        v8f;
typedef __attribute__((ext_vector_type(4)))  float        float4v;
typedef __attribute__((ext_vector_type(4)))  unsigned int uint4v;
typedef __attribute__((ext_vector_type(2)))  unsigned int uint2v;

union ABFrag { v16bf v; uint4v q[2]; };

#define S_LEN 4096
#define H_LEN 512
#define K_LEN 1024           // 2H
#define MT    64             // M tile rows per workgroup
#define MSUB  4              // 4 M-subtiles of 16 rows per wave
#define NSUB  4              // 4 N-tiles of 16 cols per wave
#define LSTR  1032           // padded LDS row stride (elements); 2064B -> 4-dword bank shift/row

__device__ __forceinline__ unsigned short f2bf(float f) {
    unsigned int u = __float_as_uint(f);
    unsigned int r = u + 0x7FFFu + ((u >> 16) & 1u);   // round-to-nearest-even
    return (unsigned short)(r >> 16);
}

// ---------- kernel 1: W_h_w fp32 -> bf16 (1 MB, L2-resident reuse) ----------
__global__ void cvt_w_kernel(const float* __restrict__ W, unsigned short* __restrict__ Wb) {
    int i = (blockIdx.x * 256 + threadIdx.x) * 4;           // 512*1024 total
    float4v f = *(const float4v*)(W + i);
    uint2v p;
    p.x = (unsigned int)f2bf(f.x) | ((unsigned int)f2bf(f.y) << 16);
    p.y = (unsigned int)f2bf(f.z) | ((unsigned int)f2bf(f.w) << 16);
    *(uint2v*)(Wb + i) = p;
}

// ---------- kernel 2: dec = decoder_output @ W_s_w^T + W_s_b  [64,512] ----------
__global__ void dec_proj_kernel(const float* __restrict__ dec,
                                const float* __restrict__ Wsw,
                                const float* __restrict__ Wsb,
                                float* __restrict__ decb) {
    int b = blockIdx.x;
    const float* dr = dec + (size_t)b * H_LEN;
    for (int h = threadIdx.x; h < H_LEN; h += 256) {
        const float* wr = Wsw + (size_t)h * H_LEN;
        float acc = Wsb[h];
        for (int k = 0; k < H_LEN; k += 4) {
            float4v a = *(const float4v*)(dr + k);
            float4v w = *(const float4v*)(wr + k);
            acc += a.x * w.x + a.y * w.y + a.z * w.z + a.w * w.w;
        }
        decb[(size_t)b * H_LEN + h] = acc;
    }
}

// ---------- kernel 3: fused enc-proj GEMM (bf16 WMMA) + tanh + v-dot ----------
// One WG = 64 M rows, 8 waves x 64 N cols. K-step: 4 A frags (LDS) x 4 B frags
// (global/L2) -> 16 WMMAs; 0.5 ds_load + 0.5 global_load per WMMA.
__global__ __launch_bounds__(256)
void fused_gemm_kernel(const float* __restrict__ enc,           // [B*S, 1024] fp32
                       const unsigned short* __restrict__ Wb,   // [512, 1024] bf16
                       const float* __restrict__ decb,          // [B, 512]
                       const float* __restrict__ Whb,           // [512]
                       const float* __restrict__ vw,            // [512]
                       float* __restrict__ scores) {            // [B*S]
    __shared__ __align__(16) unsigned short ldsA[MT * LSTR];    // ~129 KB bf16 A tile
    __shared__ float srow[MT];

    const int t  = threadIdx.x;
    const int m0 = blockIdx.x * MT;           // flattened (b,s) row base
    const int b  = m0 >> 12;                  // S = 4096, MT | S -> one b per tile

    if (t < MT) srow[t] = 0.0f;

    // ---- stage 64x1024 enc rows fp32 -> bf16 LDS (coalesced, NT to spare L2) ----
    #pragma unroll
    for (int rg = 0; rg < 4; ++rg) {
        int row = rg * 16 + (t >> 4);         // 16 threads per row
        int c0  = (t & 15) * 4;
        const float* src = enc + ((size_t)(m0 + row)) * K_LEN;
        unsigned short* dst = ldsA + row * LSTR;
        #pragma unroll
        for (int i = 0; i < 16; ++i) {
            int c = c0 + i * 64;
            float4v f = __builtin_nontemporal_load((const float4v*)(src + c));
            uint2v p;
            p.x = (unsigned int)f2bf(f.x) | ((unsigned int)f2bf(f.y) << 16);
            p.y = (unsigned int)f2bf(f.z) | ((unsigned int)f2bf(f.w) << 16);
            *(uint2v*)(dst + c) = p;
        }
    }
    __syncthreads();

    const int lane = t & 31;
    const int wave = t >> 5;                  // 8 waves x 64 columns = 512
    const int half = lane >> 4;
    const int nb   = wave * 64;
    const float* decrow = decb + (size_t)b * H_LEN;

    // A fragment base: row = lane&15; K halves per ISA 16-bit A 16x32 layout
    const unsigned short* abase = ldsA + (lane & 15) * LSTR + 8 * half;
    // B fragment base: col = nb + (lane&15); nt selects +nt*16 rows of W (imm offsets)
    const unsigned short* bbase = Wb + (size_t)(nb + (lane & 15)) * K_LEN + 8 * half;

    v8f acc[NSUB][MSUB] = {};                 // 128 VGPRs of accumulators

    #pragma unroll 1
    for (int k0 = 0; k0 < K_LEN; k0 += 32) {
        ABFrag a[MSUB], bf[NSUB];
        #pragma unroll
        for (int mt = 0; mt < MSUB; ++mt) {
            const unsigned short* arow = abase + mt * 16 * LSTR + k0;
            a[mt].q[0] = *(const uint4v*)(arow);        // K = k0 + 8*half + 0..7
            a[mt].q[1] = *(const uint4v*)(arow + 16);   // K = k0 + 16 + 8*half + 0..7
        }
        #pragma unroll
        for (int nt = 0; nt < NSUB; ++nt) {
            const unsigned short* brow = bbase + nt * 16 * K_LEN + k0;
            bf[nt].q[0] = *(const uint4v*)(brow);       // K = k0 + 8*half + 0..7
            bf[nt].q[1] = *(const uint4v*)(brow + 16);  // K = k0 + 16 + 8*half + 0..7
        }
        #pragma unroll
        for (int nt = 0; nt < NSUB; ++nt)
            #pragma unroll
            for (int mt = 0; mt < MSUB; ++mt)
                acc[nt][mt] = __builtin_amdgcn_wmma_f32_16x16x32_bf16(
                    false, a[mt].v, false, bf[nt].v, (short)0, acc[nt][mt], false, false);
    }

    // ---- epilogue: rowtot[mt][j] = sum_nt v[n]*tanh(acc + Whb[n] + dec[b,n]) ----
    float rowtot[MSUB][8];
    #pragma unroll
    for (int mt = 0; mt < MSUB; ++mt)
        #pragma unroll
        for (int j = 0; j < 8; ++j) rowtot[mt][j] = 0.0f;

    #pragma unroll
    for (int nt = 0; nt < NSUB; ++nt) {
        const int n = nb + nt * 16 + (lane & 15);
        const float bias = Whb[n] + decrow[n];
        const float vn   = vw[n];
        #pragma unroll
        for (int mt = 0; mt < MSUB; ++mt)
            #pragma unroll
            for (int j = 0; j < 8; ++j)
                rowtot[mt][j] += vn * tanhf(acc[nt][mt][j] + bias);
    }

    // reduce over the 16 columns held by the 16 lanes of this half-wave
    #pragma unroll
    for (int mt = 0; mt < MSUB; ++mt)
        #pragma unroll
        for (int j = 0; j < 8; ++j) {
            float s = rowtot[mt][j];
            s += __shfl_xor(s, 1);
            s += __shfl_xor(s, 2);
            s += __shfl_xor(s, 4);
            s += __shfl_xor(s, 8);
            rowtot[mt][j] = s;                // D row = mt*16 + j + 8*half
        }

    if ((lane & 15) == 0) {
        #pragma unroll
        for (int mt = 0; mt < MSUB; ++mt)
            #pragma unroll
            for (int j = 0; j < 8; ++j)
                atomicAdd(&srow[mt * 16 + j + 8 * half], rowtot[mt][j]);
    }
    __syncthreads();
    if (t < MT) scores[m0 + t] = srow[t];
}

// ---------- kernel 4: masked softmax over S per batch row (in-place in d_out) ----------
__global__ void softmax_kernel(float* __restrict__ scores, const int* __restrict__ mask) {
    __shared__ float red[8];
    const int b = blockIdx.x;
    const int t = threadIdx.x;
    float* row      = scores + (size_t)b * S_LEN;
    const int* mrow = mask   + (size_t)b * S_LEN;

    float loc[16];
    float mx = -INFINITY;
    #pragma unroll
    for (int i = 0; i < 16; ++i) {
        int idx = t + i * 256;
        float s = mrow[idx] ? row[idx] : -INFINITY;
        loc[i] = s;
        mx = fmaxf(mx, s);
    }
    for (int m = 1; m < 32; m <<= 1) mx = fmaxf(mx, __shfl_xor(mx, m));
    if ((t & 31) == 0) red[t >> 5] = mx;
    __syncthreads();
    mx = red[0];
    #pragma unroll
    for (int w = 1; w < 8; ++w) mx = fmaxf(mx, red[w]);
    __syncthreads();

    float sum = 0.0f;
    #pragma unroll
    for (int i = 0; i < 16; ++i) {
        loc[i] = expf(loc[i] - mx);           // exp(-inf) == 0 for masked slots
        sum += loc[i];
    }
    for (int m = 1; m < 32; m <<= 1) sum += __shfl_xor(sum, m);
    if ((t & 31) == 0) red[t >> 5] = sum;
    __syncthreads();
    sum = 0.0f;
    #pragma unroll
    for (int w = 0; w < 8; ++w) sum += red[w];
    const float inv = 1.0f / sum;
    #pragma unroll
    for (int i = 0; i < 16; ++i) row[t + i * 256] = loc[i] * inv;
}

// ---------- host launcher ----------
extern "C" void kernel_launch(void* const* d_in, const int* in_sizes, int n_in,
                              void* d_out, int out_size, void* d_ws, size_t ws_size,
                              hipStream_t stream) {
    const float* dec  = (const float*)d_in[0];   // [64,512]
    const float* enc  = (const float*)d_in[1];   // [64,4096,1024]
    const int*   mask = (const int*)d_in[2];     // [64,4096]
    const float* Whw  = (const float*)d_in[3];   // [512,1024]
    const float* Whb  = (const float*)d_in[4];   // [512]
    const float* Wsw  = (const float*)d_in[5];   // [512,512]
    const float* Wsb  = (const float*)d_in[6];   // [512]
    const float* vw   = (const float*)d_in[7];   // [1,512]
    float* out = (float*)d_out;                  // [64,4096]

    unsigned short* Wb = (unsigned short*)d_ws;                       // 1 MB bf16 weights
    float* decb = (float*)((char*)d_ws + (size_t)H_LEN * K_LEN * 2);  // 128 KB dec proj

    cvt_w_kernel   <<<(H_LEN * K_LEN) / (256 * 4), 256, 0, stream>>>(Whw, Wb);
    dec_proj_kernel<<<64, 256, 0, stream>>>(dec, Wsw, Wsb, decb);
    fused_gemm_kernel<<<(64 * S_LEN) / MT, 256, 0, stream>>>(enc, Wb, decb, Whb, vw, out);
    softmax_kernel <<<64, 256, 0, stream>>>(out, mask);
}
